// WordAttEncoder_50886772523318
// MI455X (gfx1250) — compile-verified, matrix operand order checked by reading, not run on hardware
//
#include <hip/hip_runtime.h>
#include <hip/hip_bf16.h>

// ---------------------------------------------------------------------------
// Problem constants
// ---------------------------------------------------------------------------
#define SEQ    512
#define BATCH  128
#define EMBED  300
#define KPAD   320      // EMBED padded to multiple of 32 for WMMA K
#define HID    128
#define H2     256
#define H3     384
#define NPROJ  768      // 3H for both directions side by side

typedef __attribute__((ext_vector_type(16))) __bf16 v16bf;
typedef __attribute__((ext_vector_type(8)))  float  v8f;

// fp32 -> bf16 bits, round-to-nearest-even
__device__ __forceinline__ unsigned short f2bf(float f) {
  unsigned u = __float_as_uint(f);
  unsigned r = u + 0x7FFFu + ((u >> 16) & 1u);
  return (unsigned short)(r >> 16);
}

// ---------------------------------------------------------------------------
// WMMA helpers.  Layout per CDNA5 ISA 7.12.2 (16-bit A 16x32):
//   lane L in [0,16): M = L,     VGPR q<4 -> K = 2q,2q+1 ; q>=4 -> K = 16+2(q-4)..
//   lane L in [16,32): M = L-16, same but K offset +8
// B (32x16) assumed mirrored with M->N.  C/D f32: n = lane&15, m = v + 8*(lane>=16).
// ---------------------------------------------------------------------------
__device__ __forceinline__ v8f wmma_bf16(v16bf a, v16bf b, v8f c) {
  return __builtin_amdgcn_wmma_f32_16x16x32_bf16(
      /*neg_a=*/false, a, /*neg_b=*/false, b,
      /*c_mod=*/(short)0, c, /*reuse_a=*/false, /*reuse_b=*/false);
}

// Load one 16x32 bf16 A-fragment from a row-major LDS tile (ushort elements,
// leading dim ldk).  `tile` points at (row0, k0) of the fragment.
__device__ __forceinline__ v16bf load_a_lds(const unsigned short* tile, int ldk, int lane) {
  union { v16bf v; unsigned int u[8]; } r;
  const unsigned short* p = tile + (size_t)(lane & 15) * ldk + ((lane >> 4) & 1) * 8;
#pragma unroll
  for (int q = 0; q < 8; ++q) {
    int kb = (q < 4) ? (2 * q) : (16 + 2 * (q - 4));
    r.u[q] = *(const unsigned int*)(p + kb);
  }
  return r.v;
}

// Load one 32x16 bf16 B-fragment from the pre-swizzled layout:
// element ((frag*32 + lane)*16 + h) -> one contiguous 32B load per lane.
__device__ __forceinline__ v16bf load_b_swz(const unsigned short* base, int frag, int lane) {
  return *(const v16bf*)(base + ((size_t)frag * 32 + lane) * 16);
}

// ---------------------------------------------------------------------------
// Kernel 0: convert fp32 weights into bf16 pre-swizzled B-fragment layouts.
//   wihB : K=KPAD (10 kt), N=NPROJ (48 nf)  [fwd cols 0..383, bwd cols 384..767]
//   whhF/whhB : K=128 (4 kt), N=384 (24 nf)
//   wwB  : K=256 (8 kt), N=256 (16 nf)
// ---------------------------------------------------------------------------
#define WIH_ELEMS (KPAD * NPROJ)          // 245760
#define WHH_ELEMS (HID * H3)              // 49152
#define WW_ELEMS  (H2 * H2)               // 65536
#define PREP_TOTAL (WIH_ELEMS + 2 * WHH_ELEMS + WW_ELEMS)

__global__ __launch_bounds__(256) void prep_kernel(
    const float* __restrict__ w_ih_f, const float* __restrict__ w_ih_b,
    const float* __restrict__ w_hh_f, const float* __restrict__ w_hh_b,
    const float* __restrict__ word_weight,
    unsigned short* __restrict__ wihB, unsigned short* __restrict__ whhF,
    unsigned short* __restrict__ whhB, unsigned short* __restrict__ wwB) {
  int stride = gridDim.x * blockDim.x;
  for (int e = blockIdx.x * blockDim.x + threadIdx.x; e < PREP_TOTAL; e += stride) {
    unsigned short* dst;
    int loc, Kt, mode;
    const float* src = nullptr;
    if (e < WIH_ELEMS) {
      dst = wihB; loc = e; Kt = 10; mode = 0;
    } else if (e < WIH_ELEMS + WHH_ELEMS) {
      dst = whhF; loc = e - WIH_ELEMS; Kt = 4; mode = 1; src = w_hh_f;
    } else if (e < WIH_ELEMS + 2 * WHH_ELEMS) {
      dst = whhB; loc = e - WIH_ELEMS - WHH_ELEMS; Kt = 4; mode = 1; src = w_hh_b;
    } else {
      dst = wwB; loc = e - WIH_ELEMS - 2 * WHH_ELEMS; Kt = 8; mode = 2;
    }
    int h = loc & 15;
    int lane = (loc >> 4) & 31;
    int t = loc >> 9;
    int kt = t % Kt;
    int nf = t / Kt;
    int q = h >> 1, half = h & 1;
    int k = kt * 32 + ((q < 4) ? 2 * q : 16 + 2 * (q - 4)) + ((lane >= 16) ? 8 : 0) + half;
    int n = nf * 16 + (lane & 15);
    float val;
    if (mode == 0) {
      if (k >= EMBED) val = 0.0f;
      else val = (n < H3) ? w_ih_f[n * EMBED + k] : w_ih_b[(n - H3) * EMBED + k];
    } else if (mode == 1) {
      val = src[n * HID + k];                 // gh = h @ W_hh^T -> B[k][n] = W[n][k]
    } else {
      val = word_weight[k * H2 + n];          // u = f @ W      -> B[k][n] = W[k][n]
    }
    dst[loc] = f2bf(val);
  }
}

// ---------------------------------------------------------------------------
// Kernel 1: embedding gather + input projection for both directions.
// Grid: 512 blocks, each owns 128 rows of the [65536, KPAD] x [KPAD, 768] GEMM.
// Block: 256 threads = 8 waves; wave w owns 16 rows, loops over 48 n-frags.
// LDS: bf16 A tile 128x320 (80KB) + 128 gathered row ids.
// ---------------------------------------------------------------------------
__global__ __launch_bounds__(256) void gather_project_kernel(
    const int* __restrict__ ids, const float* __restrict__ emb,
    const float* __restrict__ b_ih_f, const float* __restrict__ b_ih_b,
    const unsigned short* __restrict__ wihB,
    float* __restrict__ gi_f, float* __restrict__ gi_b) {
  extern __shared__ unsigned char smem_gp[];
  unsigned short* xA = (unsigned short*)smem_gp;           // 128*KPAD ushorts
  int* idsl = (int*)(smem_gp + 128 * KPAD * 2);            // 128 ints
  int tid = threadIdx.x;
  int m0 = blockIdx.x * 128;

  if (tid < 128) idsl[tid] = ids[m0 + tid];
  __syncthreads();

  // gather + fp32->bf16 into LDS, K padded with zeros
  for (int e = tid; e < 128 * KPAD; e += 256) {
    int r = e / KPAD;
    int k = e - r * KPAD;
    float v = 0.0f;
    if (k < EMBED) v = emb[(size_t)idsl[r] * EMBED + k];
    xA[e] = f2bf(v);
  }
  __syncthreads();

  int lane = tid & 31;
  int wave = tid >> 5;  // 8 waves -> 8 m-fragments

  // hoist this wave's 10 A-fragments (K loop) into registers
  v16bf afr[10];
#pragma unroll
  for (int kt = 0; kt < 10; ++kt)
    afr[kt] = load_a_lds(xA + (size_t)wave * 16 * KPAD + kt * 32, KPAD, lane);

  for (int nf = 0; nf < 48; ++nf) {
    v8f acc = {};
#pragma unroll
    for (int kt = 0; kt < 10; ++kt) {
      v16bf b = load_b_swz(wihB, nf * 10 + kt, lane);
      acc = wmma_bf16(afr[kt], b, acc);
    }
    int n = nf * 16 + (lane & 15);
    float bias;
    float* dst;
    int ncol;
    if (n < H3) { bias = b_ih_f[n]; dst = gi_f; ncol = n; }
    else        { bias = b_ih_b[n - H3]; dst = gi_b; ncol = n - H3; }
#pragma unroll
    for (int v = 0; v < 8; ++v) {
      int m = wave * 16 + v + ((lane >> 4) & 1) * 8;
      dst[(size_t)(m0 + m) * H3 + ncol] = acc[v] + bias;
    }
  }
}

// ---------------------------------------------------------------------------
// Kernel 2: the serial GRU recurrence (one workgroup per direction).
// 512 threads = 16 waves.  W_hh is staged ONCE into LDS via async-to-LDS
// copies (ASYNCcnt path), so the per-step critical path only touches LDS.
// Per step: stage h (regs) -> bf16 LDS; gh = h @ W_hh^T via WMMA; fused GRU
// gate elementwise; h stays in registers (stable ownership).
// LDS: hA 32KB + gh fp32 192KB + whh bf16 96KB = 320KB (workgroup max).
// ---------------------------------------------------------------------------
__global__ __launch_bounds__(512) void gru_kernel(
    const float* __restrict__ h0s,
    const float* __restrict__ b_hh_f, const float* __restrict__ b_hh_b,
    const unsigned short* __restrict__ whhF, const unsigned short* __restrict__ whhB,
    const float* __restrict__ gi_f, const float* __restrict__ gi_b,
    float* __restrict__ fout, float* __restrict__ hout) {
  extern __shared__ unsigned char smem_gru[];
  unsigned short* hA = (unsigned short*)smem_gru;                    // 32KB
  float* gh = (float*)(smem_gru + HID * BATCH * 2);                  // 192KB
  unsigned short* whhLds =
      (unsigned short*)(smem_gru + HID * BATCH * 2 + BATCH * H3 * 4);  // 96KB

  int dir = blockIdx.x;
  const float* gi = dir ? gi_b : gi_f;
  const unsigned short* whh = dir ? whhB : whhF;
  const float* bhh = dir ? b_hh_b : b_hh_f;

  int tid = threadIdx.x;           // 0..511
  int lane = tid & 31;
  int wave = tid >> 5;             // 16 waves

  // ---- stage W_hh (pre-swizzled bf16) into LDS with async copies ----------
  // 98304 bytes = 6144 x 16B transfers; 12 per thread.  Tracked by ASYNCcnt.
  {
    unsigned ldsbase = (unsigned)(uintptr_t)whhLds;   // low 32b = LDS offset
    for (int c = tid; c < WHH_ELEMS / 8; c += 512) {
      unsigned loff = ldsbase + (unsigned)c * 16u;
      unsigned long long ga = (unsigned long long)(uintptr_t)(whh + (size_t)c * 8);
      asm volatile("global_load_async_to_lds_b128 %0, %1, off"
                   :: "v"(loff), "v"(ga)
                   : "memory");
    }
    asm volatile("s_wait_asynccnt 0" ::: "memory");
  }

  // each thread owns 32 h elements: idx = i*512 + tid  (== b*128 + j)
  float hreg[32];
  const float* h0 = h0s + (size_t)dir * BATCH * HID;
#pragma unroll
  for (int i = 0; i < 32; ++i) hreg[i] = h0[i * 512 + tid];
  __syncthreads();   // whhLds visible to all waves

  // each wave owns one m-fragment for the whole kernel
  int mf = wave >> 1;             // 8 m-fragments over 16 waves
  int nf0 = (wave & 1) * 12;      // half the 24 n-fragments each

  for (int t = 0; t < SEQ; ++t) {
    int s = dir ? (SEQ - 1 - t) : t;
    const float* girow = gi + (size_t)s * BATCH * H3;
    // prefetch next step's gi rows while we compute this one
    if (t + 1 < SEQ) {
      int sn = dir ? (s - 1) : (s + 1);
      __builtin_prefetch(gi + (size_t)sn * BATCH * H3 + tid * 96, 0, 0);
    }

    // stage h as bf16 A tile (row-major [b][j])
#pragma unroll
    for (int i = 0; i < 32; ++i) hA[i * 512 + tid] = f2bf(hreg[i]);
    __syncthreads();

    // gh = h @ W_hh^T : M=128 (8 mf) x N=384 (24 nf), K=128 (4 kt)
    v16bf afr[4];
#pragma unroll
    for (int kt = 0; kt < 4; ++kt)
      afr[kt] = load_a_lds(hA + (size_t)mf * 16 * HID + kt * 32, HID, lane);

    for (int jn = 0; jn < 12; ++jn) {
      int nf = nf0 + jn;
      v8f acc = {};
#pragma unroll
      for (int kt = 0; kt < 4; ++kt) {
        v16bf b = load_b_swz(whhLds, nf * 4 + kt, lane);
        acc = wmma_bf16(afr[kt], b, acc);
      }
      int n = nf * 16 + (lane & 15);
#pragma unroll
      for (int v = 0; v < 8; ++v) {
        int m = mf * 16 + v + ((lane >> 4) & 1) * 8;
        gh[m * H3 + n] = acc[v];
      }
    }
    __syncthreads();

    // fused GRU gates; h kept in registers
#pragma unroll
    for (int i = 0; i < 32; ++i) {
      int idx = i * 512 + tid;
      int b = idx >> 7;
      int jj = idx & 127;
      const float* g = girow + b * H3;
      float i_r = g[jj], i_z = g[HID + jj], i_n = g[2 * HID + jj];
      float h_r = gh[b * H3 + jj] + bhh[jj];
      float h_z = gh[b * H3 + HID + jj] + bhh[HID + jj];
      float h_n = gh[b * H3 + 2 * HID + jj] + bhh[2 * HID + jj];
      float r = 1.0f / (1.0f + __expf(-(i_r + h_r)));
      float z = 1.0f / (1.0f + __expf(-(i_z + h_z)));
      float nn = tanhf(i_n + r * h_n);
      float hn = (1.0f - z) * nn + z * hreg[i];
      hreg[i] = hn;
      // f_output[s][b][dir*128 + jj]  (backward rows land reversed naturally)
      fout[((size_t)s * BATCH + b) * H2 + dir * HID + jj] = hn;
    }
    // no barrier needed here: next stage writes hA (all GEMM reads of hA are
    // done at barrier 2), and next GEMM's gh writes are fenced by barrier 1.
  }

  // final hidden state -> d_out h_output region
#pragma unroll
  for (int i = 0; i < 32; ++i)
    hout[(size_t)dir * BATCH * HID + i * 512 + tid] = hreg[i];
}

// ---------------------------------------------------------------------------
// Kernel 3: attention scores.  u = tanh(f @ W + b); score = tanh(u . c).
// Grid: 512 blocks x 128 rows.  GEMM M=128 N=256 K=256 via WMMA.
// LDS: bf16 f tile 128x256 (64KB) + fp32 u tile 128x256 (128KB).
// ---------------------------------------------------------------------------
__global__ __launch_bounds__(256) void attn_scores_kernel(
    const float* __restrict__ fout, const unsigned short* __restrict__ wwB,
    const float* __restrict__ word_bias, const float* __restrict__ ctx_w,
    float* __restrict__ scores) {
  extern __shared__ unsigned char smem_at[];
  unsigned short* fA = (unsigned short*)smem_at;           // 128*256
  float* su = (float*)(smem_at + 128 * H2 * 2);            // 128*256
  int tid = threadIdx.x;
  int m0 = blockIdx.x * 128;

  for (int e = tid; e < 128 * H2; e += 256) {
    int r = e >> 8;
    int k = e & 255;
    fA[e] = f2bf(fout[(size_t)(m0 + r) * H2 + k]);
  }
  __syncthreads();

  int lane = tid & 31;
  int wave = tid >> 5;  // 8 waves -> 8 m-fragments

  v16bf afr[8];
#pragma unroll
  for (int kt = 0; kt < 8; ++kt)
    afr[kt] = load_a_lds(fA + (size_t)wave * 16 * H2 + kt * 32, H2, lane);

  for (int nf = 0; nf < 16; ++nf) {
    v8f acc = {};
#pragma unroll
    for (int kt = 0; kt < 8; ++kt) {
      v16bf b = load_b_swz(wwB, nf * 8 + kt, lane);
      acc = wmma_bf16(afr[kt], b, acc);
    }
    int n = nf * 16 + (lane & 15);
    float bias = word_bias[n];
#pragma unroll
    for (int v = 0; v < 8; ++v) {
      int m = wave * 16 + v + ((lane >> 4) & 1) * 8;
      su[m * H2 + n] = tanhf(acc[v] + bias);
    }
  }
  __syncthreads();

  if (tid < 128) {
    float ssum = 0.0f;
    for (int n = 0; n < H2; ++n) ssum += su[tid * H2 + n] * ctx_w[n];
    scores[m0 + tid] = tanhf(ssum);   // row index == s*128+b
  }
}

// ---------------------------------------------------------------------------
// Kernel 4: softmax over seq + weighted pooling.  One block per batch column.
// ---------------------------------------------------------------------------
__global__ __launch_bounds__(256) void softmax_pool_kernel(
    const float* __restrict__ scores, const float* __restrict__ fout,
    float* __restrict__ pooled) {
  __shared__ float red[256];
  __shared__ float attn[SEQ];
  int b = blockIdx.x;
  int tid = threadIdx.x;

  float v0 = scores[tid * BATCH + b];
  float v1 = scores[(tid + 256) * BATCH + b];
  red[tid] = fmaxf(v0, v1);
  __syncthreads();
  for (int st = 128; st > 0; st >>= 1) {
    if (tid < st) red[tid] = fmaxf(red[tid], red[tid + st]);
    __syncthreads();
  }
  float mx = red[0];
  __syncthreads();

  float e0 = __expf(v0 - mx);
  float e1 = __expf(v1 - mx);
  attn[tid] = e0;
  attn[tid + 256] = e1;
  red[tid] = e0 + e1;
  __syncthreads();
  for (int st = 128; st > 0; st >>= 1) {
    if (tid < st) red[tid] += red[tid + st];
    __syncthreads();
  }
  float inv = 1.0f / red[0];
  __syncthreads();
  attn[tid] *= inv;
  attn[tid + 256] *= inv;
  __syncthreads();

  // each thread accumulates one of the 256 feature dims over 512 timesteps
  float acc = 0.0f;
  for (int s = 0; s < SEQ; ++s)
    acc += attn[s] * fout[((size_t)s * BATCH + b) * H2 + tid];
  pooled[b * H2 + tid] = acc;
}

// ---------------------------------------------------------------------------
// Launcher.  Workspace layout (needs ~270MB):
//   gi_f[512*128*384] f32 | gi_b[...] f32 | fout[512*128*256] f32 |
//   scores[512*128] f32 | wihB | whhF | whhB | wwB  (bf16, pre-swizzled)
// ---------------------------------------------------------------------------
extern "C" void kernel_launch(void* const* d_in, const int* in_sizes, int n_in,
                              void* d_out, int out_size, void* d_ws, size_t ws_size,
                              hipStream_t stream) {
  const int*   ids  = (const int*)  d_in[0];
  const float* h0   = (const float*)d_in[1];
  const float* emb  = (const float*)d_in[2];
  const float* wihf = (const float*)d_in[3];
  const float* whhf = (const float*)d_in[4];
  const float* bihf = (const float*)d_in[5];
  const float* bhhf = (const float*)d_in[6];
  const float* wihb = (const float*)d_in[7];
  const float* whhb = (const float*)d_in[8];
  const float* bihb = (const float*)d_in[9];
  const float* bhhb = (const float*)d_in[10];
  const float* ww   = (const float*)d_in[11];
  const float* wb   = (const float*)d_in[12];
  const float* cw   = (const float*)d_in[13];
  float* out = (float*)d_out;

  float* gi_f = (float*)d_ws;
  float* gi_b = gi_f + (size_t)SEQ * BATCH * H3;
  float* fout = gi_b + (size_t)SEQ * BATCH * H3;
  float* scr  = fout + (size_t)SEQ * BATCH * H2;
  unsigned short* wihB = (unsigned short*)(scr + SEQ * BATCH);
  unsigned short* whhF = wihB + WIH_ELEMS;
  unsigned short* whhB = whhF + WHH_ELEMS;
  unsigned short* wwB  = whhB + WHH_ELEMS;

  prep_kernel<<<1600, 256, 0, stream>>>(wihf, wihb, whhf, whhb, ww,
                                        wihB, whhF, whhB, wwB);

  size_t sh_gp = (size_t)128 * KPAD * 2 + 128 * sizeof(int);       // ~82KB
  gather_project_kernel<<<SEQ * BATCH / 128, 256, sh_gp, stream>>>(
      ids, emb, bihf, bihb, wihB, gi_f, gi_b);

  size_t sh_gru = (size_t)BATCH * HID * 2 + (size_t)BATCH * H3 * 4
                + (size_t)WHH_ELEMS * 2;                           // 320KB
  gru_kernel<<<2, 512, sh_gru, stream>>>(h0, bhhf, bhhb, whhF, whhB,
                                         gi_f, gi_b, fout,
                                         out + (size_t)BATCH * H2);

  size_t sh_at = (size_t)128 * H2 * 2 + (size_t)128 * H2 * 4;      // 192KB
  attn_scores_kernel<<<SEQ * BATCH / 128, 256, sh_at, stream>>>(
      fout, wwB, wb, cw, scr);

  softmax_pool_kernel<<<BATCH, 256, 0, stream>>>(scr, fout, out);
}
